// AttentionBase_12747462934792
// MI455X (gfx1250) — matrix operand order repaired and checked by reading, not compiled, not run on hardware
//
#include <hip/hip_runtime.h>
#include <hip/hip_bf16.h>

// ---------------------------------------------------------------------------
// Types / helpers
// ---------------------------------------------------------------------------
typedef __bf16 bf16_t;
typedef __attribute__((ext_vector_type(16))) __bf16 v16bf;
typedef __attribute__((ext_vector_type(8)))  __bf16 v8bf;
typedef __attribute__((ext_vector_type(4)))  __bf16 v4bf;
typedef __attribute__((ext_vector_type(8)))  float   v8f;

// 16-byte int vector in the address spaces the async-LDS builtin expects
typedef __attribute__((__vector_size__(16))) int v4i;
typedef __attribute__((address_space(1))) v4i as1_v4i;
typedef __attribute__((address_space(3))) v4i as3_v4i;

#define LOG2E 1.4426950408889634f

static __device__ __forceinline__ v8f wmma_bf16(v16bf a, v16bf b, v8f c) {
  // D = A(16x32) * B(32x16) + C, f32 accumulate
  return __builtin_amdgcn_wmma_f32_16x16x32_bf16(
      /*neg_a=*/false, a, /*neg_b=*/false, b,
      /*c_mod=*/(short)0, c, /*reuse_a=*/false, /*reuse_b=*/false);
}

// A fragment 16x32 (M x K) from row-major bf16 (row = M), leading dim ld.
// ISA layout: lanes 0-15 hold K=0..7 & 16..23; lanes 16-31 hold K=8..15 & 24..31.
static __device__ __forceinline__ v16bf load_frag_a(const bf16_t* src, int ld, int lane) {
  const int r  = lane & 15;
  const int kb = (lane >> 4) << 3;
  v8bf lo = *reinterpret_cast<const v8bf*>(src + (size_t)r * ld + kb);
  v8bf hi = *reinterpret_cast<const v8bf*>(src + (size_t)r * ld + kb + 16);
  v16bf f;
#pragma unroll
  for (int i = 0; i < 8; ++i) { f[i] = lo[i]; f[i + 8] = hi[i]; }
  return f;
}

// B fragment 32x16 (K x N) loaded from the TRANSPOSED operand stored row-major
// with rows = N (i.e. src[n][k] = B[k][n]).  Lane n = lane&15 reads K=kh..kh+7
// and kh+16..kh+23 with kh = (lane>>4)*8 (same layout as A).
static __device__ __forceinline__ v16bf load_frag_bT(const bf16_t* src, int ld, int lane) {
  const int n  = lane & 15;
  const int kh = (lane >> 4) << 3;
  v8bf lo = *reinterpret_cast<const v8bf*>(src + (size_t)n * ld + kh);
  v8bf hi = *reinterpret_cast<const v8bf*>(src + (size_t)n * ld + kh + 16);
  v16bf f;
#pragma unroll
  for (int i = 0; i < 8; ++i) { f[i] = lo[i]; f[i + 8] = hi[i]; }
  return f;
}

// Async global -> LDS copy, 16B per lane (GLOBAL_LOAD_ASYNC_TO_LDS_B128,
// tracked by ASYNCcnt, no VGPR staging).
static __device__ __forceinline__ void async_copy16(const bf16_t* g, bf16_t* l) {
  __builtin_amdgcn_global_load_async_to_lds_b128(
      (as1_v4i*)(g), (as3_v4i*)(l), /*imm_offset=*/0, /*cpol=*/0);
}

static __device__ __forceinline__ void wait_async0() {
  asm volatile("s_wait_asynccnt 0x0" ::: "memory");
}

// ---------------------------------------------------------------------------
// fp32 -> bf16 conversion (with optional scale; q gets D^-0.5 folded in)
// ---------------------------------------------------------------------------
__global__ __launch_bounds__(256) void cvt_bf16_kernel(const float* __restrict__ in,
                                                       bf16_t* __restrict__ out,
                                                       float scale, int n) {
  int i = (blockIdx.x * 256 + threadIdx.x) * 4;
  if (i + 3 < n) {
    float4 f = *reinterpret_cast<const float4*>(in + i);
    v4bf o;
    o[0] = (bf16_t)(f.x * scale);
    o[1] = (bf16_t)(f.y * scale);
    o[2] = (bf16_t)(f.z * scale);
    o[3] = (bf16_t)(f.w * scale);
    *reinterpret_cast<v4bf*>(out + i) = o;
  }
}

// ---------------------------------------------------------------------------
// Fused flash attention, double-buffered async K/V staging.
// qbf/kbf/vbf: [B,S,H*D] bf16 (q pre-scaled by D^-0.5).  Output [B,S,H*D] bf16.
// WG = 64 query rows of one (b,h); wave = one 16-row query tile.
// Per 32-key block: 4 WMMA (QK^T) + 1 WMMA (rowsum via ones) + 4 WMMA (PV).
// ---------------------------------------------------------------------------
__global__ __launch_bounds__(128) void attn_fused_kernel(
    const bf16_t* __restrict__ qbf, const bf16_t* __restrict__ kbf,
    const bf16_t* __restrict__ vbf, const unsigned char* __restrict__ mask,
    bf16_t* __restrict__ attnbf) {
  constexpr int S = 2048, C = 1024, H = 16, D = 64;
  constexpr int NBLK = S / 32;
  constexpr float NEG = -1.0e12f;

  __shared__ __attribute__((aligned(16))) bf16_t Ksh[2][32 * 64];   // K block row-major [key][d]
  __shared__ __attribute__((aligned(16))) bf16_t Vraw[2][32 * 64];  // V block row-major (async staging)
  __shared__ __attribute__((aligned(16))) bf16_t Vt[2][64 * 32];    // V block transposed [d][key]
  __shared__ __attribute__((aligned(16))) bf16_t Pw[4][16 * 32];    // per-wave P tile [q][k]

  const int tid  = threadIdx.x;
  const int lane = tid & 31;
  const int w    = tid >> 5;

  const int qb = blockIdx.x % (S / 64);
  const int h  = (blockIdx.x / (S / 64)) % H;
  const int b  = blockIdx.x / ((S / 64) * H);
  const int q0 = qb * 64 + w * 16;

  const bf16_t* kbase = kbf + (size_t)b * S * C + h * D;
  const bf16_t* vbase = vbf + (size_t)b * S * C + h * D;

  // Q tile: 16 rows x 64 dims -> two A fragments (dims 0..31, 32..63)
  const bf16_t* qrow = qbf + ((size_t)b * S + q0) * C + h * D;
  const v16bf Qa0 = load_frag_a(qrow, C, lane);
  const v16bf Qa1 = load_frag_a(qrow + 32, C, lane);

  // all-ones B fragment: rowsum(P) = P x ones via one WMMA
  v16bf onesB;
#pragma unroll
  for (int i = 0; i < 16; ++i) onesB[i] = (bf16_t)1.0f;

  float m[8], l[8];
  v8f acc[4];
#pragma unroll
  for (int r = 0; r < 8; ++r) { m[r] = -3.0e38f; l[r] = 0.0f; }
#pragma unroll
  for (int n = 0; n < 4; ++n) acc[n] = v8f{0.f, 0.f, 0.f, 0.f, 0.f, 0.f, 0.f, 0.f};

  const int kc = lane & 15;  // key column within 16-wide score tile

  // Cooperative async issue of one 32x64 K block + V block (16B per lane,
  // 2 passes x 128 threads covers the 4KB tile).
  auto issue_block = [&](int blk, int buf) {
#pragma unroll
    for (int pass = 0; pass < 2; ++pass) {
      int e  = pass * 128 + tid;
      int kk = e >> 3;
      int d8 = (e & 7) * 8;
      const bf16_t* gk = kbase + (size_t)(blk * 32 + kk) * C + d8;
      const bf16_t* gv = vbase + (size_t)(blk * 32 + kk) * C + d8;
      async_copy16(gk, &Ksh[buf][kk * 64 + d8]);
      async_copy16(gv, &Vraw[buf][kk * 64 + d8]);
    }
  };

  issue_block(0, 0);  // prologue

  for (int blk = 0; blk < NBLK; ++blk) {
    const int buf = blk & 1;
    const int kb  = blk * 32;

    wait_async0();     // this wave's copies for `buf` have landed
    __syncthreads();   // every wave's copies have landed; prior reads of buf^1 done

    // LDS->LDS transpose of V block (Vraw[buf] -> Vt[buf])
#pragma unroll
    for (int pass = 0; pass < 2; ++pass) {
      int e  = pass * 128 + tid;
      int kk = e >> 3;
      int d8 = (e & 7) * 8;
      v8bf vv = *reinterpret_cast<const v8bf*>(&Vraw[buf][kk * 64 + d8]);
#pragma unroll
      for (int j = 0; j < 8; ++j) Vt[buf][(d8 + j) * 32 + kk] = vv[j];
    }

    // prefetch next block into the other buffer (overlaps with compute below)
    if (blk + 1 < NBLK) issue_block(blk + 1, buf ^ 1);

    __syncthreads();   // Vt[buf] ready

    // --- scores: S = Q * K^T, two 16x16 tiles covering 32 keys -------------
    v8f s0 = v8f{0.f, 0.f, 0.f, 0.f, 0.f, 0.f, 0.f, 0.f};
    v8f s1 = s0;
    s0 = wmma_bf16(Qa0, load_frag_bT(&Ksh[buf][0], 64, lane), s0);
    s0 = wmma_bf16(Qa1, load_frag_bT(&Ksh[buf][0] + 32, 64, lane), s0);
    s1 = wmma_bf16(Qa0, load_frag_bT(&Ksh[buf][16 * 64], 64, lane), s1);
    s1 = wmma_bf16(Qa1, load_frag_bT(&Ksh[buf][16 * 64] + 32, 64, lane), s1);

    const float bias0 = mask[(size_t)b * S + kb + kc] ? 0.f : NEG;
    const float bias1 = mask[(size_t)b * S + kb + 16 + kc] ? 0.f : NEG;

    // --- online softmax: row max via width-16 butterflies ------------------
    float sc[8];
#pragma unroll
    for (int r = 0; r < 8; ++r) {
      float a0 = s0[r] + bias0;
      float a1 = s1[r] + bias1;
      float mx = fmaxf(a0, a1);
#pragma unroll
      for (int off = 1; off < 16; off <<= 1) mx = fmaxf(mx, __shfl_xor(mx, off, 16));
      float mnew = fmaxf(m[r], mx);
      sc[r] = exp2f((m[r] - mnew) * LOG2E);
      float p0 = exp2f((a0 - mnew) * LOG2E);
      float p1 = exp2f((a1 - mnew) * LOG2E);
      m[r] = mnew;
#pragma unroll
      for (int n = 0; n < 4; ++n) acc[n][r] *= sc[r];
      // stash P (C-layout -> LDS) for re-load in A-layout
      int qr = (lane < 16) ? r : r + 8;
      Pw[w][qr * 32 + kc]      = (bf16_t)p0;
      Pw[w][qr * 32 + 16 + kc] = (bf16_t)p1;
    }

    const v16bf Pa = load_frag_a(&Pw[w][0], 32, lane);

    // --- row sums with one WMMA: R[q,*] = sum_k P[q,k] ---------------------
    v8f R = wmma_bf16(Pa, onesB,
                      v8f{0.f, 0.f, 0.f, 0.f, 0.f, 0.f, 0.f, 0.f});
#pragma unroll
    for (int r = 0; r < 8; ++r) l[r] = l[r] * sc[r] + R[r];

    // --- PV: acc += P(16x32) * V(32x64) ------------------------------------
#pragma unroll
    for (int n = 0; n < 4; ++n)
      acc[n] = wmma_bf16(Pa, load_frag_bT(&Vt[buf][(n * 16) * 32], 32, lane), acc[n]);
  }

  // --- epilogue: normalize, store bf16 [B,S,H*D] -----------------------------
#pragma unroll
  for (int r = 0; r < 8; ++r) {
    int qr = (lane < 16) ? r : r + 8;
    float inv = 1.0f / l[r];
    size_t o = ((size_t)b * S + q0 + qr) * C + h * D + kc;
#pragma unroll
    for (int n = 0; n < 4; ++n) attnbf[o + n * 16] = (bf16_t)(acc[n][r] * inv);
  }
}

// ---------------------------------------------------------------------------
// Projection: out[N,C] = x[N,C] @ W^T   (y[i,j] = sum_c x[i,c] * W[j,c])
// W row-major is already the "transposed" operand -> contiguous B-fragments.
// WG = 64x64 output tile, wave = 16 rows x 64 cols.
// ---------------------------------------------------------------------------
__global__ __launch_bounds__(128) void proj_kernel(const bf16_t* __restrict__ xbf,
                                                   const bf16_t* __restrict__ wbf,
                                                   float* __restrict__ out) {
  constexpr int C = 1024;
  const int tid  = threadIdx.x;
  const int lane = tid & 31;
  const int w    = tid >> 5;
  const int rb = (blockIdx.x & 63) * 64 + w * 16;  // 4096 rows / 64
  const int jb = (blockIdx.x >> 6) * 64;           // 1024 cols / 64

  v8f acc[4];
#pragma unroll
  for (int n = 0; n < 4; ++n) acc[n] = v8f{0.f, 0.f, 0.f, 0.f, 0.f, 0.f, 0.f, 0.f};

  for (int kb = 0; kb < C; kb += 32) {
    const v16bf A = load_frag_a(xbf + (size_t)rb * C + kb, C, lane);
#pragma unroll
    for (int n = 0; n < 4; ++n) {
      const v16bf Bf = load_frag_bT(wbf + (size_t)(jb + n * 16) * C + kb, C, lane);
      acc[n] = wmma_bf16(A, Bf, acc[n]);
    }
  }

  const int jc = lane & 15;
#pragma unroll
  for (int r = 0; r < 8; ++r) {
    int qr = (lane < 16) ? r : r + 8;
#pragma unroll
    for (int n = 0; n < 4; ++n)
      out[(size_t)(rb + qr) * C + jb + n * 16 + jc] = acc[n][r];
  }
}

// ---------------------------------------------------------------------------
// Launch
// ---------------------------------------------------------------------------
extern "C" void kernel_launch(void* const* d_in, const int* in_sizes, int n_in,
                              void* d_out, int out_size, void* d_ws, size_t ws_size,
                              hipStream_t stream) {
  (void)in_sizes; (void)n_in; (void)out_size; (void)ws_size;
  constexpr int B = 2, S = 2048, C = 1024;
  constexpr size_t NQKV = (size_t)B * S * C;   // 4M elements
  constexpr size_t NW   = (size_t)C * C;       // 1M elements

  const float* q = (const float*)d_in[0];
  const float* k = (const float*)d_in[1];
  const float* v = (const float*)d_in[2];
  const unsigned char* mask = (const unsigned char*)d_in[3];
  const float* W = (const float*)d_in[4];

  bf16_t* qbf = (bf16_t*)d_ws;
  bf16_t* kbf = qbf + NQKV;
  bf16_t* vbf = kbf + NQKV;
  bf16_t* wbf = vbf + NQKV;
  bf16_t* abf = wbf + NW;   // attention output, bf16 [B,S,C]

  const float qscale = 0.125f;  // D^-0.5, D = 64
  cvt_bf16_kernel<<<NQKV / 1024, 256, 0, stream>>>(q, qbf, qscale, (int)NQKV);
  cvt_bf16_kernel<<<NQKV / 1024, 256, 0, stream>>>(k, kbf, 1.0f, (int)NQKV);
  cvt_bf16_kernel<<<NQKV / 1024, 256, 0, stream>>>(v, vbf, 1.0f, (int)NQKV);
  cvt_bf16_kernel<<<NW / 1024, 256, 0, stream>>>(W, wbf, 1.0f, (int)NW);

  attn_fused_kernel<<<B * 16 * (S / 64), 128, 0, stream>>>(qbf, kbf, vbf, mask, abf);
  proj_kernel<<<(4096 / 64) * (C / 64), 128, 0, stream>>>(abf, wbf, (float*)d_out);
}
